// QuantumClassifier_386547057080
// MI455X (gfx1250) — matrix operand order repaired and checked by reading, not compiled, not run on hardware
//
#include <hip/hip_runtime.h>
#include <hip/hip_fp16.h>

typedef __attribute__((ext_vector_type(16))) _Float16 v16h;
typedef __attribute__((ext_vector_type(8)))  _Float16 v8h;
typedef __attribute__((ext_vector_type(8)))  float    v8f;

#define M_MODES   10
#define N_PHOT    5
#define HIDDEN    100
#define IN_DIM    784
#define N_CLASSES 10
#define BATCH     128
#define N_STATES  2002
#define K1P       800   // IN_DIM padded to multiple of 32
#define N1P       112   // HIDDEN padded to multiple of 16
#define K2P       2016  // N_STATES padded to multiple of 32
#define N2P       16    // N_CLASSES padded to multiple of 16

// ---------------------------------------------------------------------------
// f32 -> f16 conversion with zero padding into [rows_pad x cols_pad]
// ---------------------------------------------------------------------------
__global__ void cvt_pad_f16(const float* __restrict__ src, _Float16* __restrict__ dst,
                            int rows, int cols, int cols_pad, int rows_pad)
{
    int i = blockIdx.x * blockDim.x + threadIdx.x;
    int total = rows_pad * cols_pad;
    if (i >= total) return;
    int rp = i / cols_pad;
    int cp = i - rp * cols_pad;
    float v = (rp < rows && cp < cols) ? src[(size_t)rp * cols + cp] : 0.0f;
    dst[i] = (_Float16)v;
}

// ---------------------------------------------------------------------------
// WMMA f16 GEMM: C[M x N] = A[Mp x Kp] * B[Np x Kp]^T + bias, opt sigmoid.
// One wave (32 threads) per 16x16 output tile; K-loop of v_wmma_f32_16x16x32_f16.
// A/B layouts per CDNA5 ISA 16-bit operand tables (wave32).
// ---------------------------------------------------------------------------
__global__ void gemm_wmma_f16(const _Float16* __restrict__ A, const _Float16* __restrict__ B,
                              const float* __restrict__ bias, float* __restrict__ C,
                              int M, int N, int Kp, int ldc, int act)
{
    const int lane = threadIdx.x & 31;
    const int r    = lane & 15;
    const int half = lane >> 4;
    const int tm   = blockIdx.x;
    const int tn   = blockIdx.y;

    const _Float16* arow = A + (size_t)(tm * 16 + r) * Kp;
    const _Float16* brow = B + (size_t)(tn * 16 + r) * Kp;

    v8f acc = {};
    for (int k0 = 0; k0 < Kp; k0 += 32) {
        // A 16x32 f16: lanes 0-15 -> elems0..7 = K0..7, elems8..15 = K16..23
        //              lanes16-31 -> elems0..7 = K8..15, elems8..15 = K24..31
        const _Float16* ap = arow + k0 + 8 * half;
        v8h a0 = *(const v8h*)(ap);
        v8h a1 = *(const v8h*)(ap + 16);
        v16h av = __builtin_shufflevector(a0, a1, 0, 1, 2, 3, 4, 5, 6, 7,
                                                  8, 9, 10, 11, 12, 13, 14, 15);
        // B 32x16 f16: same K order as A (lanes 0-15: K0..7,K16..23; lanes 16-31: K8..15,K24..31)
        const _Float16* bp = brow + k0 + 8 * half;
        v8h b0 = *(const v8h*)(bp);
        v8h b1 = *(const v8h*)(bp + 16);
        v16h bv = __builtin_shufflevector(b0, b1, 0, 1, 2, 3, 4, 5, 6, 7,
                                                  8, 9, 10, 11, 12, 13, 14, 15);
        acc = __builtin_amdgcn_wmma_f32_16x16x32_f16(
            /*neg_a=*/false, av, /*neg_b=*/false, bv,
            /*c_mod=*/(short)0, acc, /*reuse_a=*/false, /*reuse_b=*/false);
    }

    // D 16x16 f32: VGPR v -> M = v + 8*half, N = lane%16
#pragma unroll
    for (int v = 0; v < 8; ++v) {
        int m = tm * 16 + 8 * half + v;
        int n = tn * 16 + r;
        if (m < M && n < N) {
            float val = acc[v] + bias[n];
            if (act) val = 1.0f / (1.0f + expf(-val));
            C[(size_t)m * ldc + n] = val;
        }
    }
}

// ---------------------------------------------------------------------------
// theta[b][q] = sum_g h[b][g*10+q]
// ---------------------------------------------------------------------------
__global__ void theta_kernel(const float* __restrict__ h, float* __restrict__ theta)
{
    int i = blockIdx.x * blockDim.x + threadIdx.x;
    if (i >= BATCH * M_MODES) return;
    int b = i / M_MODES;
    int q = i - b * M_MODES;
    float s = 0.0f;
#pragma unroll
    for (int g = 0; g < HIDDEN / M_MODES; ++g)
        s += h[b * HIDDEN + g * M_MODES + q];
    theta[i] = s;
}

// ---------------------------------------------------------------------------
// Enumerate combinations_with_replacement(range(10), 5) in lex order and the
// inverse of the multiplicity-factorial normalizer. 2002 states; 1 thread.
// ---------------------------------------------------------------------------
__global__ void build_states(int* __restrict__ states, float* __restrict__ invfact)
{
    if (blockIdx.x != 0 || threadIdx.x != 0) return;
    int st[N_PHOT];
    for (int i = 0; i < N_PHOT; ++i) st[i] = 0;
    const float factTab[6] = {1.f, 1.f, 2.f, 6.f, 24.f, 120.f};
    for (int s = 0; s < N_STATES; ++s) {
        float f = 1.0f;
        int run = 1;
        states[s * N_PHOT + 0] = st[0];
        for (int i = 1; i < N_PHOT; ++i) {
            states[s * N_PHOT + i] = st[i];
            if (st[i] == st[i - 1]) { run++; }
            else { f *= factTab[run]; run = 1; }
        }
        f *= factTab[run];
        invfact[s] = 1.0f / f;
        // lexicographic successor
        int i = N_PHOT - 1;
        while (i >= 0 && st[i] == M_MODES - 1) --i;
        if (i < 0) break;
        int v = st[i] + 1;
        for (int j = i; j < N_PHOT; ++j) st[j] = v;
    }
}

// ---------------------------------------------------------------------------
// A[b][p][c] = sum_q WR[p][q] * e^{i theta[b][q]} * WL[q][2c],  c = 0..4
// ---------------------------------------------------------------------------
__global__ void unitary_kernel(const float* __restrict__ wl_re, const float* __restrict__ wl_im,
                               const float* __restrict__ wr_re, const float* __restrict__ wr_im,
                               const float* __restrict__ theta,
                               float* __restrict__ Are, float* __restrict__ Aim)
{
    int b = blockIdx.x;
    int t = threadIdx.x;
    if (t >= M_MODES * N_PHOT) return;
    int p = t / N_PHOT;
    int c = t - p * N_PHOT;
    int col = 2 * c;
    float accr = 0.0f, acci = 0.0f;
#pragma unroll
    for (int q = 0; q < M_MODES; ++q) {
        float th = theta[b * M_MODES + q];
        float cs = cosf(th), sn = sinf(th);
        float wrr = wr_re[p * M_MODES + q], wri = wr_im[p * M_MODES + q];
        float t1r = wrr * cs - wri * sn;
        float t1i = wrr * sn + wri * cs;
        float wlr = wl_re[q * M_MODES + col], wli = wl_im[q * M_MODES + col];
        accr += t1r * wlr - t1i * wli;
        acci += t1r * wli + t1i * wlr;
    }
    Are[((size_t)b * M_MODES + p) * N_PHOT + c] = accr;
    Aim[((size_t)b * M_MODES + p) * N_PHOT + c] = acci;
}

// ---------------------------------------------------------------------------
// prob[b][s] = |perm(A[b][STATES[s], :])|^2 / FACT[s]
// Ryser over 32 row subsets via Gray code; fully unrolled so the 5x5 complex
// tile and the 5 running column sums stay in VGPRs.
// ---------------------------------------------------------------------------
__global__ void perm_kernel(const float* __restrict__ Are, const float* __restrict__ Aim,
                            const int* __restrict__ states, const float* __restrict__ invfact,
                            float* __restrict__ prob)
{
    int tid = blockIdx.x * blockDim.x + threadIdx.x;
    if (tid >= BATCH * N_STATES) return;
    int b = tid / N_STATES;
    int s = tid - b * N_STATES;

    float mre[N_PHOT][N_PHOT], mim[N_PHOT][N_PHOT];
#pragma unroll
    for (int n = 0; n < N_PHOT; ++n) {
        int row = states[s * N_PHOT + n];
        const float* ar = Are + ((size_t)b * M_MODES + row) * N_PHOT;
        const float* ai = Aim + ((size_t)b * M_MODES + row) * N_PHOT;
#pragma unroll
        for (int j = 0; j < N_PHOT; ++j) { mre[n][j] = ar[j]; mim[n][j] = ai[j]; }
    }

    float sre[N_PHOT] = {}, sim[N_PHOT] = {};
    float pre = 0.0f, pim = 0.0f;
    int pc = 0;
#pragma unroll
    for (int k = 1; k < 32; ++k) {
        // Gray code: bit ctz(k) flips between g(k-1) and g(k)
        int n = __builtin_ctz(k);          // constant-folded per unrolled iter
        int g = k ^ (k >> 1);
        int setbit = (g >> n) & 1;
        float flip = setbit ? 1.0f : -1.0f;
        pc += setbit ? 1 : -1;
#pragma unroll
        for (int j = 0; j < N_PHOT; ++j) {
            sre[j] += flip * mre[n][j];
            sim[j] += flip * mim[n][j];
        }
        float pr = sre[0], pi = sim[0];
#pragma unroll
        for (int j = 1; j < N_PHOT; ++j) {
            float nr = pr * sre[j] - pi * sim[j];
            float ni = pr * sim[j] + pi * sre[j];
            pr = nr; pi = ni;
        }
        // signs[k] = (-1)^(5 - popcount) == +1 when popcount odd
        float sg = (pc & 1) ? 1.0f : -1.0f;
        pre += sg * pr;
        pim += sg * pi;
    }
    prob[tid] = (pre * pre + pim * pim) * invfact[s];
}

// ---------------------------------------------------------------------------
// Host-side launch sequence (all on `stream`, graph-capture safe)
// ---------------------------------------------------------------------------
extern "C" void kernel_launch(void* const* d_in, const int* in_sizes, int n_in,
                              void* d_out, int out_size, void* d_ws, size_t ws_size,
                              hipStream_t stream)
{
    (void)in_sizes; (void)n_in; (void)out_size; (void)ws_size;

    const float* x     = (const float*)d_in[0];
    const float* W1    = (const float*)d_in[1];
    const float* b1    = (const float*)d_in[2];
    const float* wl_re = (const float*)d_in[3];
    const float* wl_im = (const float*)d_in[4];
    const float* wr_re = (const float*)d_in[5];
    const float* wr_im = (const float*)d_in[6];
    const float* W2    = (const float*)d_in[7];
    const float* b2    = (const float*)d_in[8];
    float* out = (float*)d_out;

    char* wsb = (char*)d_ws;
    size_t off = 0;
    auto alloc = [&](size_t bytes) -> char* {
        off = (off + 255) & ~(size_t)255;
        char* p = wsb + off;
        off += bytes;
        return p;
    };

    _Float16* xh      = (_Float16*)alloc((size_t)BATCH * K1P * 2);
    _Float16* w1h     = (_Float16*)alloc((size_t)N1P * K1P * 2);
    float*    hbuf    = (float*)   alloc((size_t)BATCH * HIDDEN * 4);
    float*    theta   = (float*)   alloc((size_t)BATCH * M_MODES * 4);
    float*    Are     = (float*)   alloc((size_t)BATCH * M_MODES * N_PHOT * 4);
    float*    Aim     = (float*)   alloc((size_t)BATCH * M_MODES * N_PHOT * 4);
    int*      states  = (int*)     alloc((size_t)N_STATES * N_PHOT * 4);
    float*    invfact = (float*)   alloc((size_t)N_STATES * 4);
    float*    prob    = (float*)   alloc((size_t)BATCH * N_STATES * 4);
    _Float16* probh   = (_Float16*)alloc((size_t)BATCH * K2P * 2);
    _Float16* w2h     = (_Float16*)alloc((size_t)N2P * K2P * 2);

    int n;

    // stage f16 operands for GEMM1
    n = BATCH * K1P;
    cvt_pad_f16<<<(n + 255) / 256, 256, 0, stream>>>(x, xh, BATCH, IN_DIM, K1P, BATCH);
    n = N1P * K1P;
    cvt_pad_f16<<<(n + 255) / 256, 256, 0, stream>>>(W1, w1h, HIDDEN, IN_DIM, K1P, N1P);

    // h = sigmoid(x @ W1^T + b1)   (WMMA)
    dim3 g1(BATCH / 16, N1P / 16);
    gemm_wmma_f16<<<g1, 32, 0, stream>>>(xh, w1h, b1, hbuf, BATCH, HIDDEN, K1P, HIDDEN, 1);

    // theta
    n = BATCH * M_MODES;
    theta_kernel<<<(n + 255) / 256, 256, 0, stream>>>(hbuf, theta);

    // Fock states + normalizers (compile-time constants of the reference)
    build_states<<<1, 1, 0, stream>>>(states, invfact);

    // A[b] = (WR diag(e^{i theta}) WL)[:, 0:10:2]
    unitary_kernel<<<BATCH, 64, 0, stream>>>(wl_re, wl_im, wr_re, wr_im, theta, Are, Aim);

    // permanents -> probabilities
    n = BATCH * N_STATES;
    perm_kernel<<<(n + 255) / 256, 256, 0, stream>>>(Are, Aim, states, invfact, prob);

    // stage f16 operands for GEMM2
    n = BATCH * K2P;
    cvt_pad_f16<<<(n + 255) / 256, 256, 0, stream>>>(prob, probh, BATCH, N_STATES, K2P, BATCH);
    n = N2P * K2P;
    cvt_pad_f16<<<(n + 255) / 256, 256, 0, stream>>>(W2, w2h, N_CLASSES, N_STATES, K2P, N2P);

    // out = prob @ W2^T + b2   (WMMA)
    dim3 g2(BATCH / 16, N2P / 16);
    gemm_wmma_f16<<<g2, 32, 0, stream>>>(probh, w2h, b2, out, BATCH, N_CLASSES, K2P, N_CLASSES, 0);
}
